// PhasedDeltaNet_52218212385508
// MI455X (gfx1250) — compile-verified, matrix operand order checked by reading, not run on hardware
//
#include <hip/hip_runtime.h>
#include <hip/hip_bf16.h>

#define HIDDEN    2048
#define NUM_HEADS 6
#define HEAD_V    512
#define VALUE_DIM 3072
#define SEQ       2048
#define BATCH     2
#define MROWS     (BATCH * SEQ)   // 4096

// GEMM block tiling
#define BM 64
#define BN 256
#define BK 32

typedef __attribute__((ext_vector_type(16))) __bf16 v16bf;
typedef __attribute__((ext_vector_type(8)))  float  v8f;
typedef __attribute__((ext_vector_type(4)))  unsigned int v4u;
typedef __attribute__((ext_vector_type(4)))  int    v4i;
typedef __attribute__((ext_vector_type(8)))  int    v8i;

union FragBF { v16bf v; uint4 q[2]; };

// ---------------------------------------------------------------------------
// Tensor Data Mover: 2D tile load Global->LDS.
// D# per CDNA5 ISA ch.8: group0 = {count/type/addr}, group1 = {dims/strides}.
// tile = tile_rows x 32 bf16 elements, row stride = Kelems, contiguous in LDS.
// ---------------------------------------------------------------------------
__device__ __forceinline__ void tdm_load_2d(uint64_t gaddr, uint32_t lds_addr,
                                            uint32_t tile_rows, uint32_t Kelems)
{
  v4u g0 = {};
  g0[0] = 1u;                                     // count=1, user descriptor
  g0[1] = lds_addr;                               // LDS byte address
  g0[2] = (uint32_t)gaddr;                        // global_addr[31:0]
  g0[3] = (uint32_t)((gaddr >> 32) & 0x01ffffffu) // global_addr[56:32]
        | (2u << 30);                             // type=2 ("image")
  v8i g1 = {};
  g1[0] = (int)(1u << 16);                        // data_size=1 (2 bytes/elem)
  g1[1] = (int)((Kelems & 0xffffu) << 16);        // tensor_dim0[15:0]
  g1[2] = (int)((Kelems >> 16) |                  // tensor_dim0[31:16]
                ((tile_rows & 0xffffu) << 16));   // tensor_dim1[15:0]
  g1[3] = (int)(((tile_rows >> 16) & 0xffffu) |   // tensor_dim1[31:16]
                ((uint32_t)BK << 16));            // tile_dim0 = 32
  g1[4] = (int)(tile_rows & 0xffffu);             // tile_dim1 (tile_dim2=0)
  g1[5] = (int)Kelems;                            // tensor_dim0_stride[31:0]
  g1[6] = 0;                                      // stride0 hi | stride1 lo
  g1[7] = 0;
  v4i gz = {};
#if defined(__clang_major__) && (__clang_major__ >= 23)
  v8i gz8 = {};
  __builtin_amdgcn_tensor_load_to_lds(g0, g1, gz, gz, gz8, 0);
#else
  __builtin_amdgcn_tensor_load_to_lds(g0, g1, gz, gz, 0);
#endif
}

// ---------------------------------------------------------------------------
// bf16 WMMA GEMM with TDM->LDS double-buffered staging.
// C[M,N] = A[M,K] * B[K,N], B given transposed (Bt[N,K]).
// Block = 256 threads (8 waves, 2x4); block tile 64x256; wave tile 32x64;
// K step 32: per wave 8 x v_wmma_f32_16x16x32_bf16 fed by 12 ds_load_b128.
// ---------------------------------------------------------------------------
__global__ __launch_bounds__(256) void gemm_bf16_wmma(
    const __hip_bfloat16* __restrict__ A,   // M x K row-major
    const __hip_bfloat16* __restrict__ Bt,  // N x K row-major
    float* __restrict__ C, int M, int N, int K)
{
  __shared__ __align__(16) __hip_bfloat16 smA[2][BM * BK];
  __shared__ __align__(16) __hip_bfloat16 smB[2][BN * BK];

  const int tid  = threadIdx.x;
  const int w    = tid >> 5;
  const int lane = tid & 31;
  const int sel  = (lane >> 4) & 1;   // K chunk select per CDNA5 bf16 layout
  const int r    = lane & 15;
  const int wm   = w & 1;             // 2 waves along M
  const int wn   = w >> 1;            // 4 waves along N
  const int m0   = blockIdx.y * BM;
  const int n0   = blockIdx.x * BN;

  const bool issuer = (__builtin_amdgcn_readfirstlane(w) == 0);
  const uint64_t Abase = (uint64_t)(uintptr_t)(A  + (size_t)m0 * K);
  const uint64_t Bbase = (uint64_t)(uintptr_t)(Bt + (size_t)n0 * K);
  const uint32_t ldsA0 = (uint32_t)(uintptr_t)&smA[0][0];
  const uint32_t ldsA1 = (uint32_t)(uintptr_t)&smA[1][0];
  const uint32_t ldsB0 = (uint32_t)(uintptr_t)&smB[0][0];
  const uint32_t ldsB1 = (uint32_t)(uintptr_t)&smB[1][0];

  v8f acc[2][4] = {};

  if (issuer) {                       // prime buffer 0 (TDM ignores EXEC; branch is uniform)
    tdm_load_2d(Abase, ldsA0, BM, (uint32_t)K);
    tdm_load_2d(Bbase, ldsB0, BN, (uint32_t)K);
  }

  int buf = 0;
  for (int kb = 0; kb < K; kb += BK) {
    __syncthreads();                  // all reads of buf^1 from prev iter done
    if (issuer) {
      if (kb + BK < K) {
        const uint64_t off = (uint64_t)(kb + BK) * 2u;
        tdm_load_2d(Abase + off, buf ? ldsA0 : ldsA1, BM, (uint32_t)K);
        tdm_load_2d(Bbase + off, buf ? ldsB0 : ldsB1, BN, (uint32_t)K);
        __builtin_amdgcn_s_wait_tensorcnt(2);   // current slab landed
      } else {
        __builtin_amdgcn_s_wait_tensorcnt(0);
      }
    }
    __syncthreads();                  // slab in buf is visible to all waves

    const uint4* As = reinterpret_cast<const uint4*>(&smA[buf][0]);
    const uint4* Bs = reinterpret_cast<const uint4*>(&smB[buf][0]);
    FragBF a[2], b[4];
    #pragma unroll
    for (int i = 0; i < 2; ++i) {
      const int row = wm * 32 + i * 16 + r;     // LDS row stride = 4 uint4
      a[i].q[0] = As[row * 4 + sel];
      a[i].q[1] = As[row * 4 + sel + 2];
    }
    #pragma unroll
    for (int j = 0; j < 4; ++j) {
      const int row = wn * 64 + j * 16 + r;
      b[j].q[0] = Bs[row * 4 + sel];
      b[j].q[1] = Bs[row * 4 + sel + 2];
    }
    #pragma unroll
    for (int i = 0; i < 2; ++i)
      #pragma unroll
      for (int j = 0; j < 4; ++j)
        acc[i][j] = __builtin_amdgcn_wmma_f32_16x16x32_bf16(
            false, a[i].v, false, b[j].v, (short)0, acc[i][j], false, false);
    buf ^= 1;
  }

  #pragma unroll
  for (int i = 0; i < 2; ++i) {
    const int rm = m0 + wm * 32 + i * 16 + 8 * sel;   // C/D: VGPR g -> row g (+8 hi lanes)
    #pragma unroll
    for (int j = 0; j < 4; ++j) {
      float* cp = C + (size_t)rm * N + n0 + wn * 64 + j * 16 + r;
      #pragma unroll
      for (int g = 0; g < 8; ++g)
        cp[(size_t)g * N] = acc[i][j][g];
    }
  }
}

// ---------------------------------------------------------------------------
// fp32 -> bf16 elementwise convert
// ---------------------------------------------------------------------------
__global__ void f32_to_bf16(const float* __restrict__ in,
                            __hip_bfloat16* __restrict__ out, size_t n)
{
  size_t i = (size_t)blockIdx.x * blockDim.x + threadIdx.x;
  if (i < n) out[i] = __float2bfloat16(in[i]);
}

// ---------------------------------------------------------------------------
// Transpose + convert: W[K,N] fp32 -> Wt[N,K] bf16 (32x32 LDS tiles)
// ---------------------------------------------------------------------------
__global__ __launch_bounds__(256) void transpose_bf16(
    const float* __restrict__ W, __hip_bfloat16* __restrict__ Wt, int K, int N)
{
  __shared__ float tile[32][33];
  const int kt = blockIdx.y * 32;
  const int nt = blockIdx.x * 32;
  const int tx = threadIdx.x & 31;
  const int ty = threadIdx.x >> 5;
  #pragma unroll
  for (int i = 0; i < 32; i += 8)
    tile[ty + i][tx] = W[(size_t)(kt + ty + i) * N + nt + tx];
  __syncthreads();
  #pragma unroll
  for (int i = 0; i < 32; i += 8)
    Wt[(size_t)(nt + ty + i) * K + kt + tx] = __float2bfloat16(tile[tx][ty + i]);
}

// ---------------------------------------------------------------------------
// alpha/beta: sigmoid(X @ Wa), sigmoid(X @ Wb).  One wave per row.
// ---------------------------------------------------------------------------
__global__ __launch_bounds__(32) void ab_kernel(
    const float* __restrict__ X, const float* __restrict__ Wa,
    const float* __restrict__ Wb, float* __restrict__ alpha, float* __restrict__ beta)
{
  const int row  = blockIdx.x;
  const int lane = threadIdx.x;
  const float* x = X + (size_t)row * HIDDEN;
  float pa[NUM_HEADS] = {}, pb[NUM_HEADS] = {};
  for (int k = lane; k < HIDDEN; k += 32) {
    const float xv = x[k];
    const float* wa = Wa + (size_t)k * NUM_HEADS;
    const float* wb = Wb + (size_t)k * NUM_HEADS;
    #pragma unroll
    for (int h = 0; h < NUM_HEADS; ++h) {
      pa[h] = fmaf(xv, wa[h], pa[h]);
      pb[h] = fmaf(xv, wb[h], pb[h]);
    }
  }
  #pragma unroll
  for (int h = 0; h < NUM_HEADS; ++h) {
    float a = pa[h], b = pb[h];
    for (int off = 16; off; off >>= 1) {
      a += __shfl_xor(a, off, 32);
      b += __shfl_xor(b, off, 32);
    }
    if (lane == 0) {
      alpha[(size_t)row * NUM_HEADS + h] = 1.f / (1.f + __expf(-a));
      beta [(size_t)row * NUM_HEADS + h] = 1.f / (1.f + __expf(-b));
    }
  }
}

// ---------------------------------------------------------------------------
// Fused causal conv (K=4) + phased-delta recurrence.
// Complex form: u_t = a_t*u_{t-1} + b_t*e^{i(psi-theta)}*v_t, out = u_t.
// One block per (b,h); one thread per complex channel pair; serial over T.
// ---------------------------------------------------------------------------
__global__ __launch_bounds__(256) void scan_kernel(
    const float* __restrict__ q0, const float* __restrict__ k0,
    const float* __restrict__ v0, const float* __restrict__ alpha,
    const float* __restrict__ beta, const float* __restrict__ conv_q,
    const float* __restrict__ conv_k, const float* __restrict__ conv_v,
    float* __restrict__ u)
{
  const int bh = blockIdx.x;
  const int b  = bh / NUM_HEADS;
  const int h  = bh % NUM_HEADS;
  const int d  = threadIdx.x;              // pair index 0..255
  const int ce = h * HEAD_V + 2 * d;       // even channel within VALUE_DIM

  const float4 wq  = reinterpret_cast<const float4*>(conv_q)[ce];
  const float4 wk  = reinterpret_cast<const float4*>(conv_k)[ce];
  const float4 wve = reinterpret_cast<const float4*>(conv_v)[ce];
  const float4 wvo = reinterpret_cast<const float4*>(conv_v)[ce + 1];
  // inv_freq = PHASE_BASE^(-(2d)/D);  log2(10000) = 13.2877123795...
  const float invf = exp2f(-((float)(2 * d) / (float)HEAD_V) * 13.287712379549449f);

  float q1 = 0.f, q2 = 0.f, q3 = 0.f;
  float k1 = 0.f, k2 = 0.f, k3 = 0.f;
  float e1 = 0.f, e2 = 0.f, e3 = 0.f;
  float o1 = 0.f, o2 = 0.f, o3 = 0.f;
  float ur = 0.f, ui = 0.f;

  const size_t base = (size_t)b * SEQ * VALUE_DIM + ce;
  for (int t = 0; t < SEQ; ++t) {
    const size_t idx = base + (size_t)t * VALUE_DIM;
    const float  qc = q0[idx];
    const float  kc = k0[idx];
    const float2 vc = *reinterpret_cast<const float2*>(v0 + idx);

    const float qe = wq.x * q3 + wq.y * q2 + wq.z * q1 + wq.w * qc;
    const float ke = wk.x * k3 + wk.y * k2 + wk.z * k1 + wk.w * kc;
    const float ve = wve.x * e3 + wve.y * e2 + wve.z * e1 + wve.w * vc.x;
    const float vo = wvo.x * o3 + wvo.y * o2 + wvo.z * o1 + wvo.w * vc.y;
    q3 = q2; q2 = q1; q1 = qc;
    k3 = k2; k2 = k1; k1 = kc;
    e3 = e2; e2 = e1; e1 = vc.x;
    o3 = o2; o2 = o1; o1 = vc.y;

    const float ang = (ke - qe) * invf;     // psi - theta
    float s, c;
    __sincosf(ang, &s, &c);
    const float wr = c * ve - s * vo;
    const float wi = s * ve + c * vo;

    const int abidx = (b * SEQ + t) * NUM_HEADS + h;
    const float a  = alpha[abidx];
    const float bb = beta[abidx];
    ur = fmaf(a, ur, bb * wr);
    ui = fmaf(a, ui, bb * wi);

    *reinterpret_cast<float2*>(u + idx) = make_float2(ur, ui);
  }
}

// ---------------------------------------------------------------------------
// RMS-norm over D=512 per (b,t,h) + rms_w + SiLU(g) gate; write bf16 for Wo GEMM.
// ---------------------------------------------------------------------------
__global__ __launch_bounds__(128) void rms_gate_kernel(
    const float* __restrict__ u, const float* __restrict__ g0,
    const float* __restrict__ rms_w, __hip_bfloat16* __restrict__ outp)
{
  const int blk = blockIdx.x;             // (b*T + t) * H + h
  const int h   = blk % NUM_HEADS;
  const int bt  = blk / NUM_HEADS;
  const size_t base = (size_t)bt * VALUE_DIM + (size_t)h * HEAD_V;
  const int tid = threadIdx.x;            // 128 threads x 4 elems = 512

  const float4 uv = reinterpret_cast<const float4*>(u + base)[tid];
  float ss = uv.x * uv.x + uv.y * uv.y + uv.z * uv.z + uv.w * uv.w;
  for (int off = 16; off; off >>= 1) ss += __shfl_xor(ss, off, 32);
  __shared__ float lds[4];
  if ((tid & 31) == 0) lds[tid >> 5] = ss;
  __syncthreads();
  const float tot = lds[0] + lds[1] + lds[2] + lds[3];
  const float rms = rsqrtf(tot * (1.0f / HEAD_V) + 1e-5f);

  const float4 gv = reinterpret_cast<const float4*>(g0 + base)[tid];
  const float4 wv = reinterpret_cast<const float4*>(rms_w)[tid];
  float4 o;
  o.x = uv.x * rms * wv.x * (gv.x / (1.f + __expf(-gv.x)));
  o.y = uv.y * rms * wv.y * (gv.y / (1.f + __expf(-gv.y)));
  o.z = uv.z * rms * wv.z * (gv.z / (1.f + __expf(-gv.z)));
  o.w = uv.w * rms * wv.w * (gv.w / (1.f + __expf(-gv.w)));

  __hip_bfloat16* op = outp + base + (size_t)tid * 4;
  op[0] = __float2bfloat16(o.x);
  op[1] = __float2bfloat16(o.y);
  op[2] = __float2bfloat16(o.z);
  op[3] = __float2bfloat16(o.w);
}

// ---------------------------------------------------------------------------
// Host orchestration
// ---------------------------------------------------------------------------
extern "C" void kernel_launch(void* const* d_in, const int* in_sizes, int n_in,
                              void* d_out, int out_size, void* d_ws, size_t ws_size,
                              hipStream_t stream)
{
  const float* X      = (const float*)d_in[0];
  const float* Wq     = (const float*)d_in[1];
  const float* Wk     = (const float*)d_in[2];
  const float* Wv     = (const float*)d_in[3];
  const float* Wa     = (const float*)d_in[4];
  const float* Wb     = (const float*)d_in[5];
  const float* Wg     = (const float*)d_in[6];
  const float* Wo     = (const float*)d_in[7];
  const float* conv_q = (const float*)d_in[8];
  const float* conv_k = (const float*)d_in[9];
  const float* conv_v = (const float*)d_in[10];
  const float* rms_w  = (const float*)d_in[11];
  float* out = (float*)d_out;

  char* ws = (char*)d_ws;
  size_t off = 0;
  auto take = [&](size_t bytes) -> char* {
    char* p = ws + off;
    off += (bytes + 255) & ~(size_t)255;
    return p;
  };

  __hip_bfloat16* Xbf  = (__hip_bfloat16*)take((size_t)MROWS * HIDDEN * 2);
  __hip_bfloat16* Wtq  = (__hip_bfloat16*)take((size_t)VALUE_DIM * HIDDEN * 2);
  __hip_bfloat16* Wtk  = (__hip_bfloat16*)take((size_t)VALUE_DIM * HIDDEN * 2);
  __hip_bfloat16* Wtv  = (__hip_bfloat16*)take((size_t)VALUE_DIM * HIDDEN * 2);
  __hip_bfloat16* Wtg  = (__hip_bfloat16*)take((size_t)VALUE_DIM * HIDDEN * 2);
  __hip_bfloat16* Wto  = (__hip_bfloat16*)take((size_t)HIDDEN * VALUE_DIM * 2);
  float* q0    = (float*)take((size_t)MROWS * VALUE_DIM * 4);
  float* k0    = (float*)take((size_t)MROWS * VALUE_DIM * 4);
  float* v0    = (float*)take((size_t)MROWS * VALUE_DIM * 4);
  float* g0    = (float*)take((size_t)MROWS * VALUE_DIM * 4);
  float* u     = (float*)take((size_t)MROWS * VALUE_DIM * 4);
  float* alpha = (float*)take((size_t)MROWS * NUM_HEADS * 4);
  float* beta  = (float*)take((size_t)MROWS * NUM_HEADS * 4);
  __hip_bfloat16* outp = (__hip_bfloat16*)take((size_t)MROWS * VALUE_DIM * 2);
  (void)ws_size; (void)in_sizes; (void)n_in; (void)out_size;

  // 1. convert activations to bf16
  {
    size_t n = (size_t)MROWS * HIDDEN;
    f32_to_bf16<<<(unsigned)((n + 255) / 256), 256, 0, stream>>>(X, Xbf, n);
  }
  // 2. transpose+convert weights (W[K,N] -> Wt[N,K] bf16)
  transpose_bf16<<<dim3(VALUE_DIM / 32, HIDDEN / 32), 256, 0, stream>>>(Wq, Wtq, HIDDEN, VALUE_DIM);
  transpose_bf16<<<dim3(VALUE_DIM / 32, HIDDEN / 32), 256, 0, stream>>>(Wk, Wtk, HIDDEN, VALUE_DIM);
  transpose_bf16<<<dim3(VALUE_DIM / 32, HIDDEN / 32), 256, 0, stream>>>(Wv, Wtv, HIDDEN, VALUE_DIM);
  transpose_bf16<<<dim3(VALUE_DIM / 32, HIDDEN / 32), 256, 0, stream>>>(Wg, Wtg, HIDDEN, VALUE_DIM);
  transpose_bf16<<<dim3(HIDDEN / 32, VALUE_DIM / 32), 256, 0, stream>>>(Wo, Wto, VALUE_DIM, HIDDEN);
  // 3. big WMMA GEMMs (TDM-staged): q, k, v, g
  dim3 gg(VALUE_DIM / BN, MROWS / BM);
  gemm_bf16_wmma<<<gg, 256, 0, stream>>>(Xbf, Wtq, q0, MROWS, VALUE_DIM, HIDDEN);
  gemm_bf16_wmma<<<gg, 256, 0, stream>>>(Xbf, Wtk, k0, MROWS, VALUE_DIM, HIDDEN);
  gemm_bf16_wmma<<<gg, 256, 0, stream>>>(Xbf, Wtv, v0, MROWS, VALUE_DIM, HIDDEN);
  gemm_bf16_wmma<<<gg, 256, 0, stream>>>(Xbf, Wtg, g0, MROWS, VALUE_DIM, HIDDEN);
  // 4. alpha / beta gates
  ab_kernel<<<MROWS, 32, 0, stream>>>(X, Wa, Wb, alpha, beta);
  // 5. fused conv + recurrence (serial over T, parallel over B*H*D/2 chains)
  scan_kernel<<<BATCH * NUM_HEADS, 256, 0, stream>>>(q0, k0, v0, alpha, beta,
                                                     conv_q, conv_k, conv_v, u);
  // 6. RMS + SiLU gate -> bf16
  rms_gate_kernel<<<MROWS * NUM_HEADS, 128, 0, stream>>>(u, g0, rms_w, outp);
  // 7. output projection -> d_out (fp32)
  gemm_bf16_wmma<<<dim3(HIDDEN / BN, MROWS / BM), 256, 0, stream>>>(
      outp, Wto, out, MROWS, HIDDEN, VALUE_DIM);
}
